// MultiGCN_17205638988571
// MI455X (gfx1250) — matrix operand order repaired and checked by reading, not compiled, not run on hardware
//
#include <hip/hip_runtime.h>
#include <hip/hip_bf16.h>

// ---------------------------------------------------------------------------
// Types for CDNA5 WMMA
// ---------------------------------------------------------------------------
typedef __bf16 bhalf_t;
typedef __attribute__((ext_vector_type(16))) __bf16 v16bf;
typedef __attribute__((ext_vector_type(8)))  float  v8f;

__device__ __forceinline__ bhalf_t f2bf(float x) {
    union { float f; unsigned u; } a; a.f = x;
    unsigned r = (a.u + 0x7FFFu + ((a.u >> 16) & 1u)) >> 16;   // RNE
    union { unsigned short s; bhalf_t b; } o; o.s = (unsigned short)r;
    return o.b;
}
__device__ __forceinline__ unsigned pack2bf(float x, float y) {
    union { float f; unsigned u; } a, b; a.f = x; b.f = y;
    unsigned lo = (a.u + 0x7FFFu + ((a.u >> 16) & 1u)) >> 16;
    unsigned hi = (b.u + 0x7FFFu + ((b.u >> 16) & 1u)) >> 16;
    return (lo & 0xFFFFu) | (hi << 16);
}

// ---------------------------------------------------------------------------
// Adjacency build: A[b][dst][src] = sum_e dis[src]*dis[dst]  (+ self loops)
// ---------------------------------------------------------------------------
template <int NN, int NE>
__global__ void buildA_kernel(const int* __restrict__ edges, float* __restrict__ Aout) {
    const int b = blockIdx.x, tid = threadIdx.x;
    __shared__ float deg[NN];
    __shared__ float dis[NN];
    float* A = Aout + (size_t)b * NN * NN;
    const int* src = edges + (size_t)b * 2 * NE;
    const int* dst = src + NE;
    for (int i = tid; i < NN * NN; i += blockDim.x) A[i] = 0.f;
    for (int i = tid; i < NN; i += blockDim.x) deg[i] = 1.f;   // self loop
    __syncthreads();
    for (int e = tid; e < NE; e += blockDim.x) {
        int d = dst[e]; if (d >= 0 && d < NN) atomicAdd(&deg[d], 1.f);
    }
    __syncthreads();
    for (int i = tid; i < NN; i += blockDim.x) dis[i] = rsqrtf(deg[i]);
    __syncthreads();
    for (int e = tid; e < NE; e += blockDim.x) {
        int s = src[e], d = dst[e];
        if (s >= 0 && s < NN && d >= 0 && d < NN)
            atomicAdd(&A[d * NN + s], dis[s] * dis[d]);
    }
    for (int i = tid; i < NN; i += blockDim.x)
        atomicAdd(&A[i * NN + i], dis[i] * dis[i]);
}

// ---------------------------------------------------------------------------
// bf16 WMMA GEMM with optional split-K over gridDim.z.
// C[M,N] = A[M,K] @ B[K,N] (+bias[N] when gridDim.z==1), f32 in/out.
// 256 threads = 8 waves; tile 128x128; k-step 32; wave 2x4 -> 64x32 sub-tile.
// Interior blocks: register-staged software pipeline, branch-free inner loop.
// ---------------------------------------------------------------------------
#define TILE_K  32
#define LSTRIDE 40   // bf16 elems per LDS row (80 B; keeps b128 frag loads 16B-aligned)

union FragU { uint4 q[2]; v16bf v; };

__device__ __forceinline__ v16bf load_fragA(const bhalf_t* base, int lane) {
    // A 16x32 bf16: lanes 0-15 row=l, K 0..7 | 16..23 ; lanes 16-31 row=l-16, K 8..15 | 24..31
    int row = lane & 15;
    int kh  = (lane >> 4) << 3;
    FragU u;
    u.q[0] = *reinterpret_cast<const uint4*>(base + row * LSTRIDE + kh);
    u.q[1] = *reinterpret_cast<const uint4*>(base + row * LSTRIDE + kh + 16);
    return u.v;
}
__device__ __forceinline__ v16bf load_fragB(const bhalf_t* base, int lane) {
    // B 32x16 bf16 (LDS holds B transposed [n][k]): lanes 0-15 n=l K 0..7 | 16..23; lanes 16-31 n=l-16 K 8..15 | 24..31
    int col = lane & 15;
    int kh  = (lane >> 4) << 3;
    FragU u;
    u.q[0] = *reinterpret_cast<const uint4*>(base + col * LSTRIDE + kh);
    u.q[1] = *reinterpret_cast<const uint4*>(base + col * LSTRIDE + kh + 16);
    return u.v;
}

__global__ __launch_bounds__(256)
void gemm_bf16_kernel(const float* __restrict__ A, const float* __restrict__ Bm,
                      const float* __restrict__ bias, float* __restrict__ C,
                      int M, int N, int K) {
    __shared__ __align__(16) bhalf_t lA[128 * LSTRIDE];
    __shared__ __align__(16) bhalf_t lB[128 * LSTRIDE];

    const int tid  = threadIdx.x;
    const int lane = tid & 31;
    const int wid  = tid >> 5;
    const int wm   = wid >> 2;           // 0..1 -> 64-row slab
    const int wn   = wid & 3;            // 0..3 -> 32-col slab
    const int mBase = blockIdx.y * 128;
    const int nBase = blockIdx.x * 128;

    // split-K slice
    const int nKtot = (K + TILE_K - 1) / TILE_K;
    const int nz    = (int)gridDim.z;
    const int per   = (nKtot + nz - 1) / nz;
    const int kt0   = (int)blockIdx.z * per;
    const int kt1   = min(nKtot, kt0 + per);
    float* Cw = (nz > 1) ? (C + (size_t)blockIdx.z * (size_t)M * N) : C;
    const bool addBias = (nz == 1) && (bias != nullptr);

    v8f acc[4][2];
    #pragma unroll
    for (int i = 0; i < 4; ++i)
        #pragma unroll
        for (int j = 0; j < 2; ++j)
            #pragma unroll
            for (int r = 0; r < 8; ++r) acc[i][j][r] = 0.f;

    // staging thread mapping (fast path)
    const int ar  = tid >> 1;            // A: row 0..127
    const int ah  = (tid & 1) * 16;      // A: k half 0/16
    const int fkp = tid >> 4;            // B: k pair 0..15
    const int fn0 = (tid & 15) * 8;      // B: n offset 0..120

    const bool interior = (mBase + 128 <= M) && (nBase + 128 <= N);
    const int lastFull = K / TILE_K;     // tiles [0,lastFull) need no k-guard

    if (interior) {
        // ---- software-pipelined fast path: global->regs (k+1) overlaps WMMA (k) ----
        float av[16], b0[8], b1[8];
        const float* Arow = A + (size_t)(mBase + ar) * K + ah;

        auto gload = [&](int kt) {
            const int kBase = kt * TILE_K;
            const float* s  = Arow + kBase;
            const float* r0 = Bm + (size_t)(kBase + 2 * fkp) * N + nBase + fn0;
            if (kt < lastFull) {
                #pragma unroll
                for (int i = 0; i < 8; ++i) {
                    float2 t = reinterpret_cast<const float2*>(s)[i];
                    av[2*i] = t.x; av[2*i+1] = t.y;
                }
                #pragma unroll
                for (int i = 0; i < 4; ++i) {
                    float2 t0 = reinterpret_cast<const float2*>(r0)[i];
                    float2 t1 = reinterpret_cast<const float2*>(r0 + N)[i];
                    b0[2*i] = t0.x; b0[2*i+1] = t0.y;
                    b1[2*i] = t1.x; b1[2*i+1] = t1.y;
                }
            } else {   // single k-tail tile
                #pragma unroll
                for (int i = 0; i < 16; ++i) av[i] = (kBase + ah + i < K) ? s[i] : 0.f;
                const bool ok0 = (kBase + 2 * fkp    ) < K;
                const bool ok1 = (kBase + 2 * fkp + 1) < K;
                #pragma unroll
                for (int i = 0; i < 8; ++i) {
                    b0[i] = ok0 ? r0[i]     : 0.f;
                    b1[i] = ok1 ? r0[N + i] : 0.f;
                }
            }
        };
        auto lstore = [&]() {
            unsigned p[8];
            #pragma unroll
            for (int i = 0; i < 8; ++i) p[i] = pack2bf(av[2*i], av[2*i+1]);
            uint4* d = reinterpret_cast<uint4*>(lA + ar * LSTRIDE + ah);
            d[0] = make_uint4(p[0], p[1], p[2], p[3]);
            d[1] = make_uint4(p[4], p[5], p[6], p[7]);
            #pragma unroll
            for (int i = 0; i < 8; ++i)
                *reinterpret_cast<unsigned*>(lB + (fn0 + i) * LSTRIDE + 2 * fkp) =
                    pack2bf(b0[i], b1[i]);
        };

        if (kt0 < kt1) gload(kt0);
        for (int kt = kt0; kt < kt1; ++kt) {
            lstore();                    // waits on outstanding global loads, fills LDS
            __syncthreads();
            if (kt + 1 < kt1) gload(kt + 1);   // issue next tile's loads (latency hidden)
            v16bf fb0 = load_fragB(lB + (wn * 32 + 0)  * LSTRIDE, lane);
            v16bf fb1 = load_fragB(lB + (wn * 32 + 16) * LSTRIDE, lane);
            #pragma unroll
            for (int i = 0; i < 4; ++i) {
                v16bf fa = load_fragA(lA + (wm * 64 + i * 16) * LSTRIDE, lane);
                acc[i][0] = __builtin_amdgcn_wmma_f32_16x16x32_bf16(
                    false, fa, false, fb0, (short)0, acc[i][0], false, false);
                acc[i][1] = __builtin_amdgcn_wmma_f32_16x16x32_bf16(
                    false, fa, false, fb1, (short)0, acc[i][1], false, false);
            }
            __syncthreads();
        }
    } else {
        // ---- general guarded path (edge blocks only) ----
        const int bk = tid >> 3;             // B: k 0..31
        const int bj = (tid & 7) * 16;       // B: n chunk
        for (int kt = kt0; kt < kt1; ++kt) {
            const int kBase = kt * TILE_K;
            {
                const long long row = (long long)mBase + ar;
                bhalf_t* d = lA + ar * LSTRIDE + ah;
                if (row < M) {
                    const float* s = A + row * (long long)K + kBase + ah;
                    const int kr = K - (kBase + ah);
                    #pragma unroll
                    for (int i = 0; i < 16; ++i) d[i] = f2bf(i < kr ? s[i] : 0.f);
                } else {
                    #pragma unroll
                    for (int i = 0; i < 16; ++i) d[i] = f2bf(0.f);
                }
            }
            {
                const int kk  = kBase + bk;
                const bool ok = kk < K;
                const float* s = Bm + (long long)kk * N + nBase + bj;
                #pragma unroll
                for (int i = 0; i < 16; ++i) {
                    const int n = nBase + bj + i;
                    const float v = (ok && n < N) ? s[i] : 0.f;
                    lB[(bj + i) * LSTRIDE + bk] = f2bf(v);
                }
            }
            __syncthreads();
            v16bf fb0 = load_fragB(lB + (wn * 32 + 0)  * LSTRIDE, lane);
            v16bf fb1 = load_fragB(lB + (wn * 32 + 16) * LSTRIDE, lane);
            #pragma unroll
            for (int i = 0; i < 4; ++i) {
                v16bf fa = load_fragA(lA + (wm * 64 + i * 16) * LSTRIDE, lane);
                acc[i][0] = __builtin_amdgcn_wmma_f32_16x16x32_bf16(
                    false, fa, false, fb0, (short)0, acc[i][0], false, false);
                acc[i][1] = __builtin_amdgcn_wmma_f32_16x16x32_bf16(
                    false, fa, false, fb1, (short)0, acc[i][1], false, false);
            }
            __syncthreads();
        }
    }

    // ---------------- epilogue ----------------
    #pragma unroll
    for (int i = 0; i < 4; ++i) {
        const int m0 = mBase + wm * 64 + i * 16 + ((lane >> 4) * 8);
        #pragma unroll
        for (int j = 0; j < 2; ++j) {
            const int n = nBase + wn * 32 + j * 16 + (lane & 15);
            if (n < N) {
                const float bv = addBias ? bias[n] : 0.f;
                #pragma unroll
                for (int r = 0; r < 8; ++r) {
                    const int m = m0 + r;
                    if (m < M) Cw[(long long)m * N + n] = acc[i][j][r] + bv;
                }
            }
        }
    }
}

// ---------------------------------------------------------------------------
// Split-K reduction: C[idx] = sum_z parts[z][idx] + bias[idx % N]
// ---------------------------------------------------------------------------
__global__ void reduceK_kernel(const float* __restrict__ parts, const float* __restrict__ bias,
                               float* __restrict__ C, int MN, int N, int S) {
    const int idx = blockIdx.x * 256 + threadIdx.x;
    if (idx < MN) {
        float s = bias[idx % N];
        for (int z = 0; z < S; ++z) s += parts[(size_t)z * MN + idx];
        C[idx] = s;
    }
}

// ---------------------------------------------------------------------------
// Batched aggregation: Y[b] = A[b] (NxN) @ X[b] (Nx512) + bias
// ---------------------------------------------------------------------------
template <int NN>
__global__ void agg_kernel(const float* __restrict__ Ab, const float* __restrict__ X,
                           const float* __restrict__ bias, float* __restrict__ Y) {
    const int b = blockIdx.x, tid = threadIdx.x;
    __shared__ float As[NN * NN];
    const float* Abb = Ab + (size_t)b * NN * NN;
    for (int i = tid; i < NN * NN; i += blockDim.x) As[i] = Abb[i];
    __syncthreads();
    const float* Xb = X + (size_t)b * NN * 512;
    float* Yb = Y + (size_t)b * NN * 512;
    for (int idx = tid; idx < NN * 512; idx += blockDim.x) {
        const int row = idx >> 9, f = idx & 511;
        float s = bias[f];
        #pragma unroll 4
        for (int k = 0; k < NN; ++k) s += As[row * NN + k] * Xb[k * 512 + f];
        Yb[idx] = s;
    }
}

// ---------------------------------------------------------------------------
// Attention 1: word_self = softmax(Q Q^T / sqrt(512)) @ Q -> feats[:, :14848]
// ---------------------------------------------------------------------------
__global__ void attn1_kernel(const float* __restrict__ Q, float* __restrict__ feats) {
    const int b = blockIdx.x, tid = threadIdx.x;
    __shared__ float Qs[29 * 512];
    __shared__ float S[29 * 32];
    const float* Qb = Q + (size_t)b * 29 * 512;
    for (int i = tid; i < 29 * 512; i += 256) Qs[i] = Qb[i];
    __syncthreads();
    const float scale = 0.044194173824159216f;   // 1/sqrt(512)
    for (int idx = tid; idx < 29 * 29; idx += 256) {
        const int i = idx / 29, j = idx % 29;
        float s = 0.f;
        for (int d = 0; d < 512; ++d) s += Qs[i * 512 + d] * Qs[j * 512 + d];
        S[i * 32 + j] = s * scale;
    }
    __syncthreads();
    if (tid < 29) {
        float mx = -1e30f;
        for (int j = 0; j < 29; ++j) mx = fmaxf(mx, S[tid * 32 + j]);
        float sum = 0.f;
        for (int j = 0; j < 29; ++j) { float e = expf(S[tid * 32 + j] - mx); S[tid * 32 + j] = e; sum += e; }
        const float inv = 1.f / sum;
        for (int j = 0; j < 29; ++j) S[tid * 32 + j] *= inv;
    }
    __syncthreads();
    float* fb = feats + (size_t)b * 29696;
    for (int idx = tid; idx < 29 * 512; idx += 256) {
        const int row = idx >> 9, f = idx & 511;
        float s = 0.f;
        for (int j = 0; j < 29; ++j) s += S[row * 32 + j] * Qs[j * 512 + f];
        fb[idx] = s;
    }
}

// ---------------------------------------------------------------------------
// Attention 2: word_sg = softmax(word_self O^T / sqrt(512)) @ O -> feats[:, 14848:]
// ---------------------------------------------------------------------------
__global__ void attn2_kernel(const float* __restrict__ O, float* __restrict__ feats) {
    const int b = blockIdx.x, tid = threadIdx.x;
    __shared__ float Ws[29 * 512];
    __shared__ float S[29 * 36];
    const float* wsrc = feats + (size_t)b * 29696;
    for (int i = tid; i < 29 * 512; i += 256) Ws[i] = wsrc[i];
    __syncthreads();
    const float* Ob = O + (size_t)b * 36 * 512;
    const float scale = 0.044194173824159216f;
    for (int idx = tid; idx < 29 * 36; idx += 256) {
        const int i = idx / 36, n = idx % 36;
        float s = 0.f;
        const float* op = Ob + n * 512;
        for (int d = 0; d < 512; ++d) s += Ws[i * 512 + d] * op[d];
        S[idx] = s * scale;
    }
    __syncthreads();
    if (tid < 29) {
        float mx = -1e30f;
        for (int n = 0; n < 36; ++n) mx = fmaxf(mx, S[tid * 36 + n]);
        float sum = 0.f;
        for (int n = 0; n < 36; ++n) { float e = expf(S[tid * 36 + n] - mx); S[tid * 36 + n] = e; sum += e; }
        const float inv = 1.f / sum;
        for (int n = 0; n < 36; ++n) S[tid * 36 + n] *= inv;
    }
    __syncthreads();
    float* fb = feats + (size_t)b * 29696 + 14848;
    for (int idx = tid; idx < 29 * 512; idx += 256) {
        const int row = idx >> 9, f = idx & 511;
        float s = 0.f;
        for (int n = 0; n < 36; ++n) s += S[row * 36 + n] * Ob[n * 512 + f];
        fb[idx] = s;
    }
}

// ---------------------------------------------------------------------------
// In-place row log-softmax
// ---------------------------------------------------------------------------
__global__ void lsm_kernel(float* __restrict__ out, int C) {
    const int b = blockIdx.x, tid = threadIdx.x;
    __shared__ float red[256];
    float* x = out + (size_t)b * C;
    float mx = -1e30f;
    for (int i = tid; i < C; i += 256) mx = fmaxf(mx, x[i]);
    red[tid] = mx; __syncthreads();
    for (int s = 128; s > 0; s >>= 1) { if (tid < s) red[tid] = fmaxf(red[tid], red[tid + s]); __syncthreads(); }
    mx = red[0]; __syncthreads();
    float sum = 0.f;
    for (int i = tid; i < C; i += 256) sum += expf(x[i] - mx);
    red[tid] = sum; __syncthreads();
    for (int s = 128; s > 0; s >>= 1) { if (tid < s) red[tid] += red[tid + s]; __syncthreads(); }
    const float lse = mx + logf(red[0]);
    __syncthreads();
    for (int i = tid; i < C; i += 256) x[i] = x[i] - lse;
}

// ---------------------------------------------------------------------------
// Host orchestration
// ---------------------------------------------------------------------------
extern "C" void kernel_launch(void* const* d_in, const int* in_sizes, int n_in,
                              void* d_out, int out_size, void* d_ws, size_t ws_size,
                              hipStream_t stream) {
    (void)in_sizes; (void)n_in; (void)out_size; (void)ws_size;
    const float* dep_x  = (const float*)d_in[0];
    const int*   dep_ei = (const int*)  d_in[1];
    const float* obj_x  = (const float*)d_in[2];
    const int*   obj_ei = (const int*)  d_in[3];
    const float* Wt1 = (const float*)d_in[4];  const float* bt1 = (const float*)d_in[5];
    const float* Wt2 = (const float*)d_in[6];  const float* bt2 = (const float*)d_in[7];
    const float* Ws1 = (const float*)d_in[8];  const float* bs1 = (const float*)d_in[9];
    const float* Ws2 = (const float*)d_in[10]; const float* bs2 = (const float*)d_in[11];
    const float* Wf1 = (const float*)d_in[12]; const float* bf1 = (const float*)d_in[13];
    const float* Wf2 = (const float*)d_in[14]; const float* bf2 = (const float*)d_in[15];
    float* outp = (float*)d_out;

    const size_t B = 1024;
    const int SPLIT = 4;
    float* ws    = (float*)d_ws;
    float* Adep  = ws;                      // B*29*29
    float* Aobj  = Adep  + B * 841;         // B*36*36
    float* tmp   = Aobj  + B * 1296;        // B*36*512 (XW scratch, reused 4x)
    float* Q     = tmp   + B * 18432;       // B*29*512
    float* O     = Q     + B * 14848;       // B*36*512
    float* feats = O     + B * 18432;       // B*29696
    float* lg1   = feats + B * 29696;       // B*1842
    float* parts = lg1   + B * 1842;        // SPLIT*B*1842

    buildA_kernel<29, 56 ><<<1024, 128, 0, stream>>>(dep_ei, Adep);
    buildA_kernel<36, 160><<<1024, 128, 0, stream>>>(obj_ei, Aobj);

    auto gemm = [&](const float* A, const float* Bm, const float* bias, float* C,
                    int M, int N, int K, int splitK) {
        dim3 g((N + 127) / 128, (M + 127) / 128, splitK);
        gemm_bf16_kernel<<<g, 256, 0, stream>>>(A, Bm, bias, C, M, N, K);
    };

    // text GCN
    gemm(dep_x, Wt1, nullptr, tmp, 29696, 512, 300, 1);
    agg_kernel<29><<<1024, 256, 0, stream>>>(Adep, tmp, bt1, Q);
    gemm(Q, Wt2, nullptr, tmp, 29696, 512, 512, 1);
    agg_kernel<29><<<1024, 256, 0, stream>>>(Adep, tmp, bt2, Q);

    // scene GCN
    gemm(obj_x, Ws1, nullptr, tmp, 36864, 512, 2048, 1);
    agg_kernel<36><<<1024, 256, 0, stream>>>(Aobj, tmp, bs1, O);
    gemm(O, Ws2, nullptr, tmp, 36864, 512, 512, 1);
    agg_kernel<36><<<1024, 256, 0, stream>>>(Aobj, tmp, bs2, O);

    // attention
    attn1_kernel<<<1024, 256, 0, stream>>>(Q, feats);
    attn2_kernel<<<1024, 256, 0, stream>>>(O, feats);

    // FC head with split-K (480 blocks per GEMM instead of 120)
    const int MN = 1024 * 1842;
    gemm(feats, Wf1, nullptr, parts, 1024, 1842, 29696, SPLIT);
    reduceK_kernel<<<(MN + 255) / 256, 256, 0, stream>>>(parts, bf1, lg1, MN, 1842, SPLIT);
    gemm(lg1, Wf2, nullptr, parts, 1024, 1842, 1842, SPLIT);
    reduceK_kernel<<<(MN + 255) / 256, 256, 0, stream>>>(parts, bf2, outp, MN, 1842, SPLIT);
    lsm_kernel<<<1024, 256, 0, stream>>>(outp, 1842);
}
